// EncodeProcessDecode_87033217286591
// MI455X (gfx1250) — compile-verified, matrix operand order checked by reading, not run on hardware
//
#include <hip/hip_runtime.h>
#include <hip/hip_bf16.h>

#define N_NODES   50000
#define N_EDGES   800000
#define E_TOT     (N_EDGES + N_NODES)   // self loops appended
#define N_FEAT_IN 30
#define LATENT    128
#define HEADS     4
#define HID       32
#define OUT_GAT   128
#define STEPS     10
#define N_OUT     2
#define LRELU     0.2f
#define LN_EPS    1e-5f
#define ROWS_PB   64                    // rows per GEMM block (4 x 16-row strips)

typedef __attribute__((ext_vector_type(16))) _Float16 v16h;
typedef __attribute__((ext_vector_type(8)))  _Float16 v8h;
typedef __attribute__((ext_vector_type(8)))  float    v8f;

// ---------------------------------------------------------------------------
// small helpers
// ---------------------------------------------------------------------------
__device__ inline float atomicMaxFloat(float* addr, float val) {
    int* ia = (int*)addr;
    int old = __float_as_int(*addr);
    while (__int_as_float(old) < val) {
        int assumed = old;
        old = atomicCAS(ia, assumed, __float_as_int(val));
        if (old == assumed) break;
    }
    return __int_as_float(old);
}

__device__ inline void edge_nodes(const long long* __restrict__ ei, int e, int& s, int& d) {
    if (e < N_EDGES) { s = (int)ei[e]; d = (int)ei[N_EDGES + e]; }
    else             { s = d = e - N_EDGES; }
}

// ---------------------------------------------------------------------------
// utility kernels
// ---------------------------------------------------------------------------
__global__ void fill_f32_kernel(float* __restrict__ p, int n, float v) {
    int i = blockIdx.x * blockDim.x + threadIdx.x;
    if (i < n) p[i] = v;
}

__global__ void f32_to_f16_kernel(const float* __restrict__ s, _Float16* __restrict__ d, int n) {
    int i = blockIdx.x * blockDim.x + threadIdx.x;
    if (i < n) d[i] = (_Float16)s[i];
}

// pad rows (30 -> 32) for encoder W0, same col count
__global__ void pad_rows_f16_kernel(const float* __restrict__ s, _Float16* __restrict__ d,
                                    int rows, int padRows, int cols) {
    int i = blockIdx.x * blockDim.x + threadIdx.x;
    if (i >= padRows * cols) return;
    int r = i / cols;
    d[i] = (r < rows) ? (_Float16)s[i] : (_Float16)0.0f;
}

// x [N,30] f32 -> [N,32] f16 zero-padded
__global__ void pad_x_kernel(const float* __restrict__ x, _Float16* __restrict__ d) {
    int i = blockIdx.x * blockDim.x + threadIdx.x;
    if (i >= N_NODES * 32) return;
    int n = i >> 5, k = i & 31;
    d[i] = (k < N_FEAT_IN) ? (_Float16)x[n * N_FEAT_IN + k] : (_Float16)0.0f;
}

// ---------------------------------------------------------------------------
// WMMA tile GEMM:  C[N x Kout] = act(A[N x Kin] @ W[Kin x Kout] + bias)
// A,W f16; acc f32 via v_wmma_f32_16x16x32_f16.
// block = 256 threads (8 waves): 64-row x 128-col tile; grid.y tiles Kout.
// A staged global->LDS with GLOBAL_LOAD_ASYNC_TO_LDS_B128 (ASYNCcnt path);
// W slice staged transposed into LDS (padded stride -> aligned b128 ds reads).
// ---------------------------------------------------------------------------
__global__ __launch_bounds__(256)
void gemm_wmma_kernel(const _Float16* __restrict__ A, const _Float16* __restrict__ Wg,
                      const float* __restrict__ bias,
                      float* __restrict__ C32, _Float16* __restrict__ C16,
                      int N, int Kin, int Kout, int act)
{
    __shared__ __align__(16) _Float16 sA[ROWS_PB * 136];
    __shared__ __align__(16) _Float16 sW[128 * 136];
    const int KP = Kin + 8;                       // padded LDS stride (keeps 16B align)

    const int tid  = threadIdx.x;
    const int lane = tid & 31;
    const int wave = tid >> 5;
    const int rowBase = blockIdx.x * ROWS_PB;
    const int colBase = blockIdx.y * 128;

    // ---- stage A tile (ROWS_PB x Kin) via async global->LDS b128 copies ----
    const int kin8 = Kin >> 3;
    for (int v = tid; v < ROWS_PB * kin8; v += 256) {
        int r = v / kin8, kv = v - r * kin8;
        int rg = rowBase + r;
        unsigned int la = (unsigned int)(uintptr_t)(const void*)(sA + r * KP + kv * 8);
        if (rg < N) {
            unsigned long long ga =
                (unsigned long long)(uintptr_t)(const void*)(A + (size_t)rg * Kin + kv * 8);
            asm volatile("global_load_async_to_lds_b128 %0, %1, off"
                         :: "v"(la), "v"(ga) : "memory");
        } else {
            v8h z = {};
            *(v8h*)(sA + r * KP + kv * 8) = z;    // zero-fill pad rows
        }
    }
    // ---- stage W slice (Kin x 128) transposed: sW[c*KP + k] = W[k][colBase+c] ----
    for (int v = tid; v < Kin * 16; v += 256) {
        int k = v >> 4, cv = v & 15;
        v8h val = *(const v8h*)(Wg + (size_t)k * Kout + colBase + cv * 8);
#pragma unroll
        for (int j = 0; j < 8; ++j) sW[(cv * 8 + j) * KP + k] = val[j];
    }
    asm volatile("s_wait_asynccnt 0" ::: "memory");
    __syncthreads();

    const int mr   = lane & 15;      // A row / B col within tile
    const int hs   = lane >> 4;      // K-half selector
    const int cloc = wave * 16 + mr; // column within 128-col slice
    const int cg   = colBase + cloc;
    const float bv = bias ? bias[cg] : 0.0f;

    // B fragments are invariant across row strips: prebuild per k-step on the fly
    for (int rs = 0; rs < ROWS_PB / 16; ++rs) {
        v8f acc = { bv, bv, bv, bv, bv, bv, bv, bv };
        const int ar = rs * 16 + mr;

        for (int k0 = 0; k0 < Kin; k0 += 32) {
            v8h a0 = *(const v8h*)(sA + ar * KP + k0 + hs * 8);
            v8h a1 = *(const v8h*)(sA + ar * KP + k0 + 16 + hs * 8);
            v8h b0 = *(const v8h*)(sW + cloc * KP + k0 + hs * 8);
            v8h b1 = *(const v8h*)(sW + cloc * KP + k0 + 16 + hs * 8);
            v16h af, bf;
#pragma unroll
            for (int j = 0; j < 8; ++j) {
                af[j] = a0[j]; af[j + 8] = a1[j];
                bf[j] = b0[j]; bf[j + 8] = b1[j];
            }
            acc = __builtin_amdgcn_wmma_f32_16x16x32_f16(false, af, false, bf,
                                                         (short)0, acc, false, false);
        }

        // C/D layout: VGPR i -> row i + 8*hs within the 16-row strip; col = cg
#pragma unroll
        for (int i = 0; i < 8; ++i) {
            int rg = rowBase + rs * 16 + i + 8 * hs;
            if (rg >= N) continue;
            float v = acc[i];
            if (act == 1) v = v > 0.0f ? v : 0.0f;
            size_t o = (size_t)rg * Kout + cg;
            if (C32) C32[o] = v;
            if (C16) C16[o] = (_Float16)v;
        }
    }
}

// ---------------------------------------------------------------------------
// LayerNorm(128): one wave per node (wave32), shuffle reduction
// ---------------------------------------------------------------------------
__global__ __launch_bounds__(256)
void layernorm_kernel(const float* __restrict__ in, const float* __restrict__ g,
                      const float* __restrict__ b, _Float16* __restrict__ out)
{
    int wave = threadIdx.x >> 5, lane = threadIdx.x & 31;
    int n = blockIdx.x * 8 + wave;
    if (n >= N_NODES) return;
    const float* row = in + (size_t)n * LATENT;
    float v[4], s = 0.0f, sq = 0.0f;
#pragma unroll
    for (int i = 0; i < 4; ++i) {
        v[i] = row[lane + 32 * i];
        s += v[i]; sq += v[i] * v[i];
    }
#pragma unroll
    for (int m = 16; m >= 1; m >>= 1) {
        s  += __shfl_xor(s,  m, 32);
        sq += __shfl_xor(sq, m, 32);
    }
    float mean = s * (1.0f / LATENT);
    float var  = sq * (1.0f / LATENT) - mean * mean;
    float rstd = rsqrtf(var + LN_EPS);
#pragma unroll
    for (int i = 0; i < 4; ++i) {
        int f = lane + 32 * i;
        out[(size_t)n * LATENT + f] = (_Float16)((v[i] - mean) * rstd * g[f] + b[f]);
    }
}

// ---------------------------------------------------------------------------
// GAT pieces
// ---------------------------------------------------------------------------
// a_s[n,h] = <h[n,h,:], att_src[h,:]> ; a_d likewise
__global__ void attn_node_kernel(const _Float16* __restrict__ gh,
                                 const float* __restrict__ att_src,
                                 const float* __restrict__ att_dst,
                                 float* __restrict__ a_s, float* __restrict__ a_d, int C)
{
    int idx = blockIdx.x * blockDim.x + threadIdx.x;   // n*HEADS + h
    if (idx >= N_NODES * HEADS) return;
    int n = idx >> 2, h = idx & 3;
    const _Float16* row = gh + (size_t)n * (HEADS * C) + h * C;
    float ss = 0.0f, sd = 0.0f;
    for (int c = 0; c < C; ++c) {
        float v = (float)row[c];
        ss += v * att_src[h * C + c];
        sd += v * att_dst[h * C + c];
    }
    a_s[idx] = ss; a_d[idx] = sd;
}

// pass 1: alpha = leaky_relu(a_s[src]+a_d[dst]); scatter max per dst
__global__ void edge_alpha_max_kernel(const long long* __restrict__ ei,
                                      const float* __restrict__ a_s, const float* __restrict__ a_d,
                                      float* __restrict__ alpha, float* __restrict__ mbuf)
{
    int e = blockIdx.x * blockDim.x + threadIdx.x;
    if (e >= E_TOT) return;
    int s, d; edge_nodes(ei, e, s, d);
#pragma unroll
    for (int h = 0; h < HEADS; ++h) {
        float a = a_s[s * HEADS + h] + a_d[d * HEADS + h];
        a = a > 0.0f ? a : LRELU * a;
        alpha[e * HEADS + h] = a;
        atomicMaxFloat(&mbuf[d * HEADS + h], a);
    }
}

// pass 2: e = exp(alpha - m[dst]); scatter sum per dst; alpha <- e
__global__ void edge_exp_sum_kernel(const long long* __restrict__ ei,
                                    float* __restrict__ alpha,
                                    const float* __restrict__ mbuf, float* __restrict__ sbuf)
{
    int e = blockIdx.x * blockDim.x + threadIdx.x;
    if (e >= E_TOT) return;
    int s, d; edge_nodes(ei, e, s, d);
    (void)s;
#pragma unroll
    for (int h = 0; h < HEADS; ++h) {
        float ev = __expf(alpha[e * HEADS + h] - mbuf[d * HEADS + h]);
        alpha[e * HEADS + h] = ev;
        atomicAdd(&sbuf[d * HEADS + h], ev);
    }
}

// pass 3: out[dst] += coef * h[src]  (concat: F->F ; mean: fold heads into C)
__global__ void edge_agg_kernel(const long long* __restrict__ ei,
                                const _Float16* __restrict__ gh,
                                const float* __restrict__ ebuf, const float* __restrict__ sbuf,
                                float* __restrict__ out32, int F, int C, int concat)
{
    int idx = blockIdx.x * blockDim.x + threadIdx.x;   // e*F + j
    int total = E_TOT * F;
    if (idx >= total) return;
    int e = idx / F, j = idx - e * F;
    int s, d; edge_nodes(ei, e, s, d);
    int h = j / C;
    float coef = ebuf[e * HEADS + h] / sbuf[d * HEADS + h];
    float hv = (float)gh[(size_t)s * F + j];
    int tgt = concat ? j : (j - h * C);
    atomicAdd(&out32[(size_t)d * LATENT + tgt], coef * hv);
}

// finalize: h_next = out32 (*0.25 if mean) + bias, store f16
__global__ void gat_finalize_kernel(const float* __restrict__ out32, const float* __restrict__ b,
                                    _Float16* __restrict__ hin16, int meanFlag)
{
    int idx = blockIdx.x * blockDim.x + threadIdx.x;
    if (idx >= N_NODES * LATENT) return;
    int f = idx & (LATENT - 1);
    float v = out32[idx];
    if (meanFlag) v *= (1.0f / HEADS);
    hin16[idx] = (_Float16)(v + b[f]);
}

// decoder head 128 -> 2
__global__ void dec_final_kernel(const _Float16* __restrict__ h, const float* __restrict__ W,
                                 const float* __restrict__ b, float* __restrict__ out)
{
    int n = blockIdx.x * blockDim.x + threadIdx.x;
    if (n >= N_NODES) return;
    float o0 = b[0], o1 = b[1];
    for (int k = 0; k < LATENT; ++k) {
        float v = (float)h[(size_t)n * LATENT + k];
        o0 += v * W[k * N_OUT + 0];
        o1 += v * W[k * N_OUT + 1];
    }
    out[n * N_OUT + 0] = o0;
    out[n * N_OUT + 1] = o1;
}

// ---------------------------------------------------------------------------
// host-side orchestration
// ---------------------------------------------------------------------------
static inline int ceil_div(int a, int b) { return (a + b - 1) / b; }

extern "C" void kernel_launch(void* const* d_in, const int* in_sizes, int n_in,
                              void* d_out, int out_size, void* d_ws, size_t ws_size,
                              hipStream_t stream)
{
    (void)in_sizes; (void)n_in; (void)out_size; (void)ws_size;

    const float*     x  = (const float*)d_in[0];
    const long long* ei = (const long long*)d_in[1];
    const float* encW[3] = { (const float*)d_in[2], (const float*)d_in[3], (const float*)d_in[4] };
    const float* encB[3] = { (const float*)d_in[5], (const float*)d_in[6], (const float*)d_in[7] };
    const float* ln_g = (const float*)d_in[8];
    const float* ln_b = (const float*)d_in[9];
    const float *gatW[STEPS], *gatAS[STEPS], *gatAD[STEPS], *gatB[STEPS];
    for (int i = 0; i < STEPS; ++i) {
        gatW[i]  = (const float*)d_in[10 + 4 * i + 0];
        gatAS[i] = (const float*)d_in[10 + 4 * i + 1];
        gatAD[i] = (const float*)d_in[10 + 4 * i + 2];
        gatB[i]  = (const float*)d_in[10 + 4 * i + 3];
    }
    const float* decW[3] = { (const float*)d_in[50], (const float*)d_in[51], (const float*)d_in[52] };
    const float* decB[3] = { (const float*)d_in[53], (const float*)d_in[54], (const float*)d_in[55] };

    // workspace carve-up (256B aligned regions)
    size_t off = 0;
    auto alloc = [&](size_t bytes) -> void* {
        void* r = (char*)d_ws + off;
        off = (off + bytes + 255) & ~(size_t)255;
        return r;
    };
    _Float16* w_enc0 = (_Float16*)alloc(32 * 128 * sizeof(_Float16));
    _Float16* w_enc1 = (_Float16*)alloc(128 * 128 * sizeof(_Float16));
    _Float16* w_enc2 = (_Float16*)alloc(128 * 128 * sizeof(_Float16));
    _Float16* w_gat[STEPS];
    for (int i = 0; i < STEPS; ++i) {
        int F = HEADS * ((i < STEPS - 1) ? HID : OUT_GAT);
        w_gat[i] = (_Float16*)alloc((size_t)128 * F * sizeof(_Float16));
    }
    _Float16* w_dec0 = (_Float16*)alloc(128 * 128 * sizeof(_Float16));
    _Float16* w_dec1 = (_Float16*)alloc(128 * 128 * sizeof(_Float16));
    _Float16* xpad16 = (_Float16*)alloc((size_t)N_NODES * 32 * sizeof(_Float16));
    _Float16* hin16  = (_Float16*)alloc((size_t)N_NODES * LATENT * sizeof(_Float16));
    _Float16* gh16   = (_Float16*)alloc((size_t)N_NODES * 512 * sizeof(_Float16));  // also dec tmp
    float*    out32  = (float*)alloc((size_t)N_NODES * LATENT * sizeof(float));
    float*    a_s    = (float*)alloc((size_t)N_NODES * HEADS * sizeof(float));
    float*    a_d    = (float*)alloc((size_t)N_NODES * HEADS * sizeof(float));
    float*    mbuf   = (float*)alloc((size_t)N_NODES * HEADS * sizeof(float));
    float*    sbuf   = (float*)alloc((size_t)N_NODES * HEADS * sizeof(float));
    float*    alpha  = (float*)alloc((size_t)E_TOT * HEADS * sizeof(float));

    const int TB = 256;
    auto cvt = [&](const float* s, _Float16* d, int n) {
        f32_to_f16_kernel<<<ceil_div(n, TB), TB, 0, stream>>>(s, d, n);
    };

    // ---- weight conversions (every call; deterministic, graph-capture safe) ----
    pad_rows_f16_kernel<<<ceil_div(32 * 128, TB), TB, 0, stream>>>(encW[0], w_enc0, N_FEAT_IN, 32, 128);
    cvt(encW[1], w_enc1, 128 * 128);
    cvt(encW[2], w_enc2, 128 * 128);
    for (int i = 0; i < STEPS; ++i) {
        int F = HEADS * ((i < STEPS - 1) ? HID : OUT_GAT);
        cvt(gatW[i], w_gat[i], 128 * F);
    }
    cvt(decW[0], w_dec0, 128 * 128);
    cvt(decW[1], w_dec1, 128 * 128);
    pad_x_kernel<<<ceil_div(N_NODES * 32, TB), TB, 0, stream>>>(x, xpad16);

    const int rowTiles = ceil_div(N_NODES, ROWS_PB);
    auto gemm = [&](const _Float16* A, const _Float16* W, const float* bias,
                    float* C32, _Float16* C16, int Kin, int Kout, int act) {
        dim3 grid(rowTiles, Kout / 128);
        gemm_wmma_kernel<<<grid, 256, 0, stream>>>(A, W, bias, C32, C16, N_NODES, Kin, Kout, act);
    };

    _Float16* tmpA = gh16;                                    // N x 128 f16 scratch
    _Float16* tmpB = gh16 + (size_t)N_NODES * LATENT;         // N x 128 f16 scratch

    // ---- encoder MLP + LayerNorm ----
    gemm(xpad16, w_enc0, encB[0], nullptr, tmpA, 32, 128, 1);
    gemm(tmpA,   w_enc1, encB[1], nullptr, tmpB, 128, 128, 1);
    gemm(tmpB,   w_enc2, encB[2], out32, nullptr, 128, 128, 0);
    layernorm_kernel<<<ceil_div(N_NODES, 8), 256, 0, stream>>>(out32, ln_g, ln_b, hin16);

    // ---- 10 GAT layers ----
    for (int i = 0; i < STEPS; ++i) {
        const int C = (i < STEPS - 1) ? HID : OUT_GAT;
        const int F = HEADS * C;
        const int concat = (i < STEPS - 1) ? 1 : 0;

        fill_f32_kernel<<<ceil_div(N_NODES * HEADS, TB), TB, 0, stream>>>(mbuf, N_NODES * HEADS, -1e30f);
        fill_f32_kernel<<<ceil_div(N_NODES * HEADS, TB), TB, 0, stream>>>(sbuf, N_NODES * HEADS, 0.0f);
        fill_f32_kernel<<<ceil_div(N_NODES * LATENT, TB), TB, 0, stream>>>(out32, N_NODES * LATENT, 0.0f);

        gemm(hin16, w_gat[i], nullptr, nullptr, gh16, 128, F, 0);
        attn_node_kernel<<<ceil_div(N_NODES * HEADS, TB), TB, 0, stream>>>(gh16, gatAS[i], gatAD[i], a_s, a_d, C);
        edge_alpha_max_kernel<<<ceil_div(E_TOT, TB), TB, 0, stream>>>(ei, a_s, a_d, alpha, mbuf);
        edge_exp_sum_kernel<<<ceil_div(E_TOT, TB), TB, 0, stream>>>(ei, alpha, mbuf, sbuf);
        int total = E_TOT * F;
        edge_agg_kernel<<<ceil_div(total, TB), TB, 0, stream>>>(ei, gh16, alpha, sbuf, out32, F, C, concat);
        gat_finalize_kernel<<<ceil_div(N_NODES * LATENT, TB), TB, 0, stream>>>(out32, gatB[i], hin16, concat ? 0 : 1);
    }

    // ---- decoder MLP ----
    gemm(hin16, w_dec0, decB[0], nullptr, tmpA, 128, 128, 1);
    gemm(tmpA,  w_dec1, decB[1], nullptr, tmpB, 128, 128, 1);
    dec_final_kernel<<<ceil_div(N_NODES, TB), TB, 0, stream>>>(tmpB, decW[2], decB[2], (float*)d_out);
}
